// SelfAttentivePooling_19009525252335
// MI455X (gfx1250) — matrix operand order repaired and checked by reading, not compiled, not run on hardware
//
#include <hip/hip_runtime.h>
#include <hip/hip_bf16.h>
#include <math.h>

// ---------------------------------------------------------------------------
// Self-attentive pooling, MI455X (gfx1250, wave32, WMMA + async-LDS).
//
// Roofline: 7.3 GFLOP vs ~105 MB mandatory HBM traffic (AI ~70 FLOP/B).
// The 200x128x128 hidden GEMM rides v_wmma_f32_16x16x32_f16 (f32 accum);
// the input tensor (the HBM-dominant stream) is staged to LDS with
// global_load_async_to_lds_b128 double buffering. Per-row working set
// ~158KB LDS -> 2 workgroups / WGP. Each wave owns one N-pair, so the
// compiler keeps all 8 B fragments resident in VGPRs across the M loop.
// ---------------------------------------------------------------------------

typedef __attribute__((ext_vector_type(16))) _Float16 v16h;
typedef __attribute__((ext_vector_type(8)))  float    v8f;
typedef int v4i __attribute__((vector_size(16)));   // matches builtin's int4 vec

#define B_      1024
#define L_      200
#define E_      128
#define A_      128
#define H_      4
#define LP      208          // L padded to 13 * 16
#define XPS     132          // LDS row stride (halves) for xp (b32-aligned)
#define HS      136          // LDS row stride (halves) for hidden (b128-aligned)
#define W1S     132          // LDS row stride (halves) for W1^T
#define THREADS 128          // 4 wave32s
#define CH      8            // async staging chunk = 8 rows (2 rows / wave)

// LDS layout (bytes):
//   xpS    : LP*XPS halves = 54912
//   hidS   : LP*HS  halves = 56576   (full 208 rows: no store guards)
//   w1tS   : 128*W1S halves= 33792
//   w2tS   : 512 f32       =  2048   (W2 transposed: [h][a])
//   attS   : 4*LP f32      =  3328
//   pooled : 512 f32       =  2048
//   mskS   : 256 bytes
//   stgS   : 2*CH*E f32    =  8192   (async double buffer)
#define SMEM_BYTES (54912 + 56576 + 33792 + 2048 + 3328 + 2048 + 256 + 8192)

#if defined(__has_builtin)
#if __has_builtin(__builtin_amdgcn_global_load_async_to_lds_b128) && \
    __has_builtin(__builtin_amdgcn_s_wait_asynccnt)
#define USE_ASYNC 1
#endif
#endif
#ifndef USE_ASYNC
#define USE_ASYNC 0
#endif

// Async 16B global->LDS copy. Builtin params are v4i* in AS(1) / AS(3);
// route casts through integers (LDS pointers are 32-bit offsets in AS3).
#define ASYNC_CPY16(g, l)                                                     \
    __builtin_amdgcn_global_load_async_to_lds_b128(                           \
        (__attribute__((address_space(1))) v4i*)(unsigned long long)          \
            (const void*)(g),                                                 \
        (__attribute__((address_space(3))) v4i*)(unsigned int)                \
            (unsigned long long)(const void*)(l),                             \
        0, 0)

union HalfPack { v16h h; unsigned int u[8]; };
union F2H      { _Float16 h[2]; unsigned int u; };
union H8       { uint4 v; _Float16 h[8]; };

// Branch-free tanh: 1 - 2/(e^{2x}+1). v_exp_f32 + v_rcp_f32, saturates to +-1.
__device__ __forceinline__ float tanh_fast(float x) {
    const float e2 = __expf(2.0f * x);
    return 1.0f - 2.0f * __builtin_amdgcn_rcpf(e2 + 1.0f);
}

__global__ void __launch_bounds__(THREADS)
sap_kernel(const float* __restrict__ input,
           const unsigned char* __restrict__ mask,
           const float* __restrict__ pos,
           const float* __restrict__ W1,
           const float* __restrict__ W2,
           const float* __restrict__ Wout,
           float* __restrict__ out)
{
    extern __shared__ char smem[];
    char* sp = smem;
    _Float16* xpS     = (_Float16*)sp;  sp += LP * XPS * 2;
    _Float16* hidS    = (_Float16*)sp;  sp += LP * HS  * 2;
    _Float16* w1tS    = (_Float16*)sp;  sp += 128 * W1S * 2;
    float*    w2tS    = (float*)sp;     sp += H_ * A_ * 4;
    float*    attS    = (float*)sp;     sp += H_ * LP * 4;
    float*    pooledS = (float*)sp;     sp += H_ * E_ * 4;
    unsigned char* mskS = (unsigned char*)sp; sp += 256;
    float*    stgS    = (float*)sp;     // 2 * CH * E_ floats
    (void)stgS;

    const int tid    = threadIdx.x;
    const int lane   = tid & 31;
    const int wid    = tid >> 5;
    const int laneLo = lane & 15;
    const int laneHi = lane >> 4;        // 0 for lanes 0-15, 1 for 16-31
    const int b      = blockIdx.x;

    // Pre-touch W_out (only consumed at the very end) -> global_prefetch_b8.
    for (int j = tid; j < H_ * E_; j += THREADS)
        __builtin_prefetch(Wout + (size_t)j * E_, 0, 1);

    // ---------------- Stage A: build xp (f16), W1^T (f16), W2^T, mask ------
#if USE_ASYNC
    // Zero-pad rows 200..207 of xp.
    for (int idx = tid; idx < 8 * 32; idx += THREADS) {
        const int l = 200 + (idx >> 5);
        const int q = idx & 31;
        unsigned int* dst = (unsigned int*)(xpS + l * XPS + 4 * q);
        dst[0] = 0u; dst[1] = 0u;
        if (q == 0) mskS[l] = 0;
    }
    {
        // 25 chunks of 8 rows; each wave owns 2 rows/chunk (self-contained:
        // it issues, waits on its own ASYNCcnt, and converts -- no barriers).
        const int rbase = wid * 2;
        const float* gsrc = input + (size_t)b * L_ * E_;
#pragma unroll
        for (int r2 = 0; r2 < 2; ++r2) {   // issue chunk 0
            const int row = rbase + r2;
            const float* g = gsrc + (size_t)row * E_ + lane * 4;
            float*       l = stgS + (rbase + r2) * E_ + lane * 4;
            ASYNC_CPY16(g, l);
        }
        for (int c = 0; c < 25; ++c) {
            if (c + 1 < 25) {
                const int nb = (c + 1) & 1;
#pragma unroll
                for (int r2 = 0; r2 < 2; ++r2) {
                    const int row = (c + 1) * CH + rbase + r2;
                    const float* g = gsrc + (size_t)row * E_ + lane * 4;
                    float*       l = stgS + nb * (CH * E_) + (rbase + r2) * E_ + lane * 4;
                    ASYNC_CPY16(g, l);
                }
                __builtin_amdgcn_s_wait_asynccnt(2);   // chunk c landed
            } else {
                __builtin_amdgcn_s_wait_asynccnt(0);
            }
            const float* sbuf = stgS + (c & 1) * (CH * E_);
#pragma unroll
            for (int r2 = 0; r2 < 2; ++r2) {
                const int row = c * CH + rbase + r2;
                const unsigned char mv = mask[(size_t)b * L_ + row];
                if (lane == 0) mskS[row] = mv;
                const float fm = mv ? 1.0f : 0.0f;
                const float4 v4 = *(const float4*)(sbuf + (rbase + r2) * E_ + lane * 4);
                const float4 p4 = *(const float4*)(pos + (size_t)row * E_ + lane * 4);
                F2H a, c2;
                a.h[0]  = (_Float16)(v4.x * fm + p4.x);
                a.h[1]  = (_Float16)(v4.y * fm + p4.y);
                c2.h[0] = (_Float16)(v4.z * fm + p4.z);
                c2.h[1] = (_Float16)(v4.w * fm + p4.w);
                unsigned int* dst = (unsigned int*)(xpS + row * XPS + 4 * lane);
                dst[0] = a.u; dst[1] = c2.u;
            }
        }
    }
#else
    for (int idx = tid; idx < LP * 32; idx += THREADS) {
        const int l = idx >> 5;
        const int q = idx & 31;
        unsigned int u0 = 0, u1 = 0;
        if (l < L_) {
            const unsigned char mv = mask[(size_t)b * L_ + l];
            if (q == 0) mskS[l] = mv;
            const float fm = mv ? 1.0f : 0.0f;
            const float4 in4 = *(const float4*)(input + ((size_t)b * L_ + l) * E_ + 4 * q);
            const float4 p4  = *(const float4*)(pos + (size_t)l * E_ + 4 * q);
            F2H a, c;
            a.h[0] = (_Float16)(in4.x * fm + p4.x);
            a.h[1] = (_Float16)(in4.y * fm + p4.y);
            c.h[0] = (_Float16)(in4.z * fm + p4.z);
            c.h[1] = (_Float16)(in4.w * fm + p4.w);
            u0 = a.u; u1 = c.u;
        } else if (q == 0) {
            mskS[l] = 0;
        }
        unsigned int* dst = (unsigned int*)(xpS + l * XPS + 4 * q);
        dst[0] = u0; dst[1] = u1;
    }
#endif
    // W1^T into LDS: w1t[a][e] = W1[e][a]; coalesced f32 reads, b16 scatter.
    for (int idx = tid; idx < E_ * 32; idx += THREADS) {
        const int k = idx >> 5;       // E index
        const int q = idx & 31;       // A / 4
        const float4 w4 = *(const float4*)(W1 + (size_t)k * A_ + 4 * q);
        w1tS[(4 * q + 0) * W1S + k] = (_Float16)w4.x;
        w1tS[(4 * q + 1) * W1S + k] = (_Float16)w4.y;
        w1tS[(4 * q + 2) * W1S + k] = (_Float16)w4.z;
        w1tS[(4 * q + 3) * W1S + k] = (_Float16)w4.w;
    }
    // W2^T: w2t[h][a] = W2[a][h] (contiguous rows for vectorized Stage C).
    for (int idx = tid; idx < A_ * H_; idx += THREADS)
        w2tS[(idx & 3) * A_ + (idx >> 2)] = W2[idx];
    __syncthreads();

    // ---------------- Stage B: hidden = tanh(xp @ W1) via WMMA -------------
    // 13 M-tiles x 4 N-pairs = 52 jobs, exactly 13 per wave. np = wid & 3 is
    // wave-invariant, so the 8 B fragments stay resident in VGPRs across the
    // whole M loop; only A fragments stream from LDS. 8 WMMAs / job.
    // A-frag (16x32 f16): lane m = laneLo; dword j -> K = laneHi*8 +
    //   {0,2,4,6,16,18,20,22}[j]. B-frag (32x16 f16): lane n = laneLo;
    //   dword j -> K = laneHi*8 + {0,2,4,6,16,18,20,22}[j] (one b32 from a W1^T row).
    for (int t = wid; t < 52; t += 4) {
        const int mt = t >> 2;                 // 0..12
        const int np = t & 3;                  // N-pair: tiles np and np+4
        const int am  = mt * 16 + laneLo;
        const int bn0 = np * 16 + laneLo;
        const int bn1 = (np + 4) * 16 + laneLo;
        HalfPack a[4];
#pragma unroll
        for (int kt = 0; kt < 4; ++kt) {
            const _Float16* ap = xpS + am * XPS + laneHi * 8 + kt * 32;
            a[kt].u[0] = *(const unsigned int*)(ap + 0);
            a[kt].u[1] = *(const unsigned int*)(ap + 2);
            a[kt].u[2] = *(const unsigned int*)(ap + 4);
            a[kt].u[3] = *(const unsigned int*)(ap + 6);
            a[kt].u[4] = *(const unsigned int*)(ap + 16);
            a[kt].u[5] = *(const unsigned int*)(ap + 18);
            a[kt].u[6] = *(const unsigned int*)(ap + 20);
            a[kt].u[7] = *(const unsigned int*)(ap + 22);
        }
        v8f acc0 = {0.f, 0.f, 0.f, 0.f, 0.f, 0.f, 0.f, 0.f};
        v8f acc1 = {0.f, 0.f, 0.f, 0.f, 0.f, 0.f, 0.f, 0.f};
#pragma unroll
        for (int kt = 0; kt < 4; ++kt) {
            HalfPack b0, b1;
            const _Float16* bp0 = w1tS + bn0 * W1S + laneHi * 8 + kt * 32;
            const _Float16* bp1 = w1tS + bn1 * W1S + laneHi * 8 + kt * 32;
#pragma unroll
            for (int j = 0; j < 8; ++j) {
                b0.u[j] = *(const unsigned int*)(bp0 + 2 * j + (j >= 4 ? 8 : 0));
                b1.u[j] = *(const unsigned int*)(bp1 + 2 * j + (j >= 4 ? 8 : 0));
            }
            acc0 = __builtin_amdgcn_wmma_f32_16x16x32_f16(
                false, a[kt].h, false, b0.h, (short)0, acc0, false, false);
            acc1 = __builtin_amdgcn_wmma_f32_16x16x32_f16(
                false, a[kt].h, false, b1.h, (short)0, acc1, false, false);
        }
        // D layout: VGPR r -> (M = r + 8*laneHi, N = laneLo). Fused tanh.
        // hidS has the full 208 padded rows -> unconditional stores, no
        // exec-mask branch chain in the hot loop.
        const int mrow0 = mt * 16 + laneHi * 8;
        const int nc0   = np * 16 + laneLo;
        const int nc1   = (np + 4) * 16 + laneLo;
#pragma unroll
        for (int r = 0; r < 8; ++r) {
            hidS[(mrow0 + r) * HS + nc0] = (_Float16)tanh_fast(acc0[r]);
            hidS[(mrow0 + r) * HS + nc1] = (_Float16)tanh_fast(acc1[r]);
        }
    }
    __syncthreads();

    // ---------------- Stage C: att = hidden @ W2 (vectorized b128) ---------
    for (int idx = tid; idx < L_ * H_; idx += THREADS) {
        const int m = idx >> 2;
        const int h = idx & 3;
        float acc = 0.f;
#pragma unroll 4
        for (int n = 0; n < A_; n += 8) {
            H8 hh; hh.v = *(const uint4*)(hidS + m * HS + n);     // 16B aligned
            const float4 wa = *(const float4*)(w2tS + h * A_ + n);
            const float4 wb = *(const float4*)(w2tS + h * A_ + n + 4);
            acc += (float)hh.h[0] * wa.x + (float)hh.h[1] * wa.y +
                   (float)hh.h[2] * wa.z + (float)hh.h[3] * wa.w +
                   (float)hh.h[4] * wb.x + (float)hh.h[5] * wb.y +
                   (float)hh.h[6] * wb.z + (float)hh.h[7] * wb.w;
        }
        attS[h * LP + m] = acc;
    }
    __syncthreads();

    // ---------------- Stage D: masked softmax, one wave per head -----------
    {
        const int h = wid;                     // exactly 4 waves, 4 heads
        float mx = -3.0e38f;
        for (int l = lane; l < L_; l += 32) {
            const float a = mskS[l] ? attS[h * LP + l] : -3.0e38f;
            mx = fmaxf(mx, a);
        }
        for (int o = 16; o > 0; o >>= 1) mx = fmaxf(mx, __shfl_xor(mx, o));
        float sum = 0.f;
        for (int l = lane; l < L_; l += 32) {
            const float e = mskS[l] ? __expf(attS[h * LP + l] - mx) : 0.f;
            attS[h * LP + l] = e;
            sum += e;
        }
        for (int o = 16; o > 0; o >>= 1) sum += __shfl_xor(sum, o);
        const float inv = 1.0f / sum;          // >=1 valid position guaranteed
        for (int l = lane; l < L_; l += 32) attS[h * LP + l] *= inv;
    }
    __syncthreads();

    // ---------------- Stage E: pooled = att @ x, with x = xp - pos ---------
    {
        const int e = tid;                     // E_ == THREADS
        float a0 = 0.f, a1 = 0.f, a2 = 0.f, a3 = 0.f;
        for (int l = 0; l < L_; ++l) {
            const float xv = (float)xpS[l * XPS + e] - pos[(size_t)l * E_ + e];
            a0 += attS[0 * LP + l] * xv;
            a1 += attS[1 * LP + l] * xv;
            a2 += attS[2 * LP + l] * xv;
            a3 += attS[3 * LP + l] * xv;
        }
        pooledS[0 * E_ + e] = a0;
        pooledS[1 * E_ + e] = a1;
        pooledS[2 * E_ + e] = a2;
        pooledS[3 * E_ + e] = a3;
    }
    __syncthreads();

    // ---------------- Stage F: out = flat(pooled) @ W_out ------------------
    {
        const int e = tid;
        float acc = 0.f;
        for (int j = 0; j < H_ * E_; ++j)
            acc += pooledS[j] * Wout[(size_t)j * E_ + e];   // coalesced, L2-hot
        out[(size_t)b * E_ + e] = acc;
    }
}

extern "C" void kernel_launch(void* const* d_in, const int* in_sizes, int n_in,
                              void* d_out, int out_size, void* d_ws, size_t ws_size,
                              hipStream_t stream) {
    (void)in_sizes; (void)n_in; (void)out_size; (void)d_ws; (void)ws_size;
    const float*         input = (const float*)d_in[0];
    const unsigned char* mask  = (const unsigned char*)d_in[1];   // jax bool = 1 byte
    const float*         pos   = (const float*)d_in[2];
    const float*         W1    = (const float*)d_in[3];
    const float*         W2    = (const float*)d_in[4];
    const float*         Wout  = (const float*)d_in[5];
    float*               out   = (float*)d_out;

    // ~158KB dynamic LDS (WGP supports 320KB -> 2 workgroups/WGP).
    (void)hipFuncSetAttribute((const void*)sap_kernel,
                              hipFuncAttributeMaxDynamicSharedMemorySize,
                              SMEM_BYTES);

    sap_kernel<<<dim3(B_), dim3(THREADS), SMEM_BYTES, stream>>>(
        input, mask, pos, W1, W2, Wout, out);
}
